// Model_21998822490496
// MI455X (gfx1250) — matrix operand order repaired and checked by reading, not compiled, not run on hardware
//
#include <hip/hip_runtime.h>
#include <hip/hip_bf16.h>

#define SEQ_LEN 48
#define FEAT 35
#define HID 1024
#define BATCH 4096
#define GATES (4 * HID)     // 4096
#define KIN (2 * FEAT)      // 70
#define KPAD 1120           // 1024 + 96  (70 padded to 96; multiple of 32)

#define TM 128              // batch-rows per workgroup
#define TJ 64               // hid-cols per workgroup (x4 gates)
#define BK 32

typedef __attribute__((ext_vector_type(16))) __bf16 v16bf;
typedef __attribute__((ext_vector_type(8)))  __bf16 v8bf;
typedef __attribute__((ext_vector_type(8)))  float  v8f;

static __device__ __forceinline__ __bf16 f2bf(float f) {
    union { float f; unsigned u; } v; v.f = f;
    unsigned r = v.u + 0x7FFFu + ((v.u >> 16) & 1u);   // round-to-nearest-even
    unsigned short h = (unsigned short)(r >> 16);
    __bf16 b; __builtin_memcpy(&b, &h, 2); return b;
}

// ---------------- prep: pack W_hh|W_ih into bf16 [GATES, KPAD] ----------------
__global__ void k_prep_w(const float* __restrict__ W_hh, const float* __restrict__ W_ih,
                         __bf16* __restrict__ Wbuf) {
    long long idx = (long long)blockIdx.x * blockDim.x + threadIdx.x;
    if (idx >= (long long)GATES * KPAD) return;
    int n = (int)(idx / KPAD), k = (int)(idx % KPAD);
    float v = 0.f;
    if (k < HID)              v = W_hh[(long long)n * HID + k];
    else if (k < HID + KIN)   v = W_ih[(long long)n * KIN + (k - HID)];
    Wbuf[idx] = f2bf(v);
}

__global__ void k_zero_state(float* hbuf, float* cbuf, __bf16* Abuf,
                             const float* __restrict__ b_ih, const float* __restrict__ b_hh,
                             float* bsum, float* xl, float* bce) {
    long long idx = (long long)blockIdx.x * blockDim.x + threadIdx.x;
    if (idx < (long long)BATCH * HID) { hbuf[idx] = 0.f; cbuf[idx] = 0.f; }
    if (idx < (long long)BATCH * KPAD) Abuf[idx] = f2bf(0.f);
    if (idx < GATES) bsum[idx] = b_ih[idx] + b_hh[idx];
    if (idx < 2 * SEQ_LEN) xl[idx] = 0.f;
    if (idx < 2) bce[idx] = 0.f;
}

// -------- per-timestep: gamma decay, regression x_h, imputation, A pack --------
__global__ __launch_bounds__(256)
void k_decay_impute(int t,
                    const float* __restrict__ values, const float* __restrict__ masks,
                    const float* __restrict__ deltas,
                    const float* __restrict__ td_W, const float* __restrict__ td_b,
                    const float* __restrict__ reg_W, const float* __restrict__ reg_b,
                    const float* __restrict__ hbuf, __bf16* __restrict__ Abuf,
                    float* __restrict__ imput,
                    float* __restrict__ xl_num, float* __restrict__ xl_den) {
    __shared__ float sd[FEAT];
    __shared__ float sh[HID];
    __shared__ float s_num, s_den;
    int b = blockIdx.x, tid = threadIdx.x;
    if (tid < FEAT) sd[tid] = deltas[((long long)b * SEQ_LEN + t) * FEAT + tid];
    if (tid == 0) { s_num = 0.f; s_den = 0.f; }
    __syncthreads();

    for (int u = tid; u < HID; u += 256) {
        float acc = td_b[u];
        const float* wr = td_W + (long long)u * FEAT;
        #pragma unroll
        for (int f = 0; f < FEAT; ++f) acc += sd[f] * wr[f];
        float gamma = __expf(-fmaxf(acc, 0.f));
        float hd = hbuf[(long long)b * HID + u] * gamma;
        sh[u] = hd;
        Abuf[(long long)b * KPAD + u] = f2bf(hd);   // decayed h in bf16 for WMMA
    }
    __syncthreads();

    int lane = tid & 31, wid = tid >> 5;
    for (int f = wid; f < FEAT; f += 8) {
        float p = 0.f;
        const float* wr = reg_W + (long long)f * HID;
        for (int u = lane; u < HID; u += 32) p += sh[u] * wr[u];
        #pragma unroll
        for (int off = 16; off > 0; off >>= 1) p += __shfl_down(p, off, 32);
        if (lane == 0) {
            float x_h = p + reg_b[f];
            long long base = ((long long)b * SEQ_LEN + t) * FEAT + f;
            float x = values[base], m = masks[base];
            float x_c = m * x + (1.f - m) * x_h;
            imput[base] = x_c;
            Abuf[(long long)b * KPAD + HID + f]        = f2bf(x_c);
            Abuf[(long long)b * KPAD + HID + FEAT + f] = f2bf(m);
            atomicAdd(&s_num, fabsf(x - x_h) * m);
            atomicAdd(&s_den, m);
        }
    }
    __syncthreads();
    if (tid == 0) { atomicAdd(&xl_num[t], s_num); atomicAdd(&xl_den[t], s_den); }
}

// ---- fused gates GEMM + LSTM: all 4 gate tiles per workgroup, epilogue LSTM ----
// Async DMA (global_load_async_to_lds_b128) double-buffers LDS tiles; the K-loop
// is unrolled in buffer pairs so the ping/pong LDS addresses are compile-time
// constants (no per-iteration cndmask selection).
__global__ __launch_bounds__(256)
void k_gates_gemm_lstm(const __bf16* __restrict__ Abuf, const __bf16* __restrict__ Wbuf,
                       const float* __restrict__ bsum,
                       float* __restrict__ hbuf, float* __restrict__ cbuf) {
    __shared__ __align__(32) __bf16 As[2][TM][BK];        // 2 x 8 KB
    __shared__ __align__(32) __bf16 Bs[2][4][TJ][BK];     // 2 x 16 KB (gate, j, k)
    int tid = threadIdx.x;
    int lane = tid & 31, wid = tid >> 5;
    int waveM = wid >> 2;       // 0..1 -> 64-row band
    int waveJ = wid & 3;        // 0..3 -> 16-col band of j
    int blockM = blockIdx.y * TM;
    int blockJ = blockIdx.x * TJ;
    int half = lane >> 4;       // 0/1 half-wave
    int l16  = lane & 15;

    // per-thread loop-invariant copy coordinates (16B per thread per chunk)
    const __bf16* aSrc[2];
    unsigned aLds[2][2];
    #pragma unroll
    for (int it = 0; it < 2; ++it) {
        int e = (tid + it * 256) * 8;
        int r = e / BK, c = e % BK;
        aSrc[it] = &Abuf[(long long)(blockM + r) * KPAD + c];
        aLds[0][it] = (unsigned)(size_t)&As[0][r][c];
        aLds[1][it] = (unsigned)(size_t)&As[1][r][c];
    }
    const __bf16* bSrc[4];
    unsigned bLds[2][4];
    #pragma unroll
    for (int it = 0; it < 4; ++it) {
        int e = (tid + it * 256) * 8;
        int g = e / (TJ * BK);
        int rem = e % (TJ * BK);
        int r = rem / BK, c = rem % BK;
        bSrc[it] = &Wbuf[(long long)(g * HID + blockJ + r) * KPAD + c];
        bLds[0][it] = (unsigned)(size_t)&Bs[0][g][r][c];
        bLds[1][it] = (unsigned)(size_t)&Bs[1][g][r][c];
    }

    v8f acc[4][4];              // [m-subtile][gate] = 128 VGPRs
    #pragma unroll
    for (int i = 0; i < 4; ++i)
        #pragma unroll
        for (int g = 0; g < 4; ++g)
            #pragma unroll
            for (int e = 0; e < 8; ++e) acc[i][g][e] = 0.f;

    // ---- async-issue tile k0 into LDS buffer buf (buf is a constant) ----
    auto issue_tile = [&](int k0, int buf) {
        #pragma unroll
        for (int it = 0; it < 2; ++it) {
            const __bf16* g = aSrc[it] + k0;
            asm volatile("global_load_async_to_lds_b128 %0, %1, off"
                         :: "v"(aLds[buf][it]), "v"(g) : "memory");
        }
        #pragma unroll
        for (int it = 0; it < 4; ++it) {
            const __bf16* g = bSrc[it] + k0;
            asm volatile("global_load_async_to_lds_b128 %0, %1, off"
                         :: "v"(bLds[buf][it]), "v"(g) : "memory");
        }
    };

    // ---- one pipeline step: prefetch k0+BK into buf^1, compute from buf ----
    auto step = [&](int k0, int buf) {
        if (k0 + BK < KPAD) issue_tile(k0 + BK, buf ^ 1);

        v16bf bf[4];
        #pragma unroll
        for (int g = 0; g < 4; ++g)
            bf[g] = *(const v16bf*)&Bs[buf][g][waveJ * 16 + l16][half * 16];

        #pragma unroll
        for (int mt = 0; mt < 4; ++mt) {
            int mrow = waveM * 64 + mt * 16 + l16;
            union { v16bf v; v8bf h[2]; } a;
            // A 16x32 layout (contiguous, matching B): half-wave selects K range {h*16 .. h*16+15}
            a.h[0] = *(const v8bf*)&As[buf][mrow][half * 16];
            a.h[1] = *(const v8bf*)&As[buf][mrow][half * 16 + 8];
            #pragma unroll
            for (int g = 0; g < 4; ++g)
                acc[mt][g] = __builtin_amdgcn_wmma_f32_16x16x32_bf16(
                    false, a.v, false, bf[g], (short)0, acc[mt][g], false, false);
        }

        // next buffer fully landed + all waves done reading the current one
        asm volatile("s_wait_asynccnt 0" ::: "memory");
        __syncthreads();
    };

    issue_tile(0, 0);
    asm volatile("s_wait_asynccnt 0" ::: "memory");
    __syncthreads();

    // KPAD/BK = 35 tiles: 17 statically-specialized ping/pong pairs + tail (buf 0)
    for (int k0 = 0; k0 + 2 * BK <= KPAD; k0 += 2 * BK) {
        step(k0, 0);
        step(k0 + BK, 1);
    }
    step(KPAD - BK, 0);

    // ---- fused LSTM epilogue: gates never touch memory ----
    int j = blockJ + waveJ * 16 + l16;
    float b0 = bsum[j], b1 = bsum[HID + j], b2 = bsum[2 * HID + j], b3 = bsum[3 * HID + j];
    #pragma unroll
    for (int mt = 0; mt < 4; ++mt) {
        int mbase = blockM + waveM * 64 + mt * 16 + half * 8;  // C layout: M = r + 8*half
        #pragma unroll
        for (int r = 0; r < 8; ++r) {
            long long idx = (long long)(mbase + r) * HID + j;
            float gi = acc[mt][0][r] + b0;
            float gf = acc[mt][1][r] + b1;
            float gg = acc[mt][2][r] + b2;
            float go = acc[mt][3][r] + b3;
            float i_ = 1.f / (1.f + __expf(-gi));
            float f_ = 1.f / (1.f + __expf(-gf));
            float g_ = tanhf(gg);
            float o_ = 1.f / (1.f + __expf(-go));
            float c = f_ * cbuf[idx] + i_ * g_;
            cbuf[idx] = c;
            hbuf[idx] = o_ * tanhf(c);
        }
    }
}

// ------------------- output head: logits, sigmoid, BCE partials ----------------
__global__ __launch_bounds__(256)
void k_out(const float* __restrict__ hbuf, const float* __restrict__ out_W,
           const float* __restrict__ out_b, const float* __restrict__ labels,
           const float* __restrict__ is_train, float* __restrict__ pred,
           float* __restrict__ bce_acc) {
    __shared__ float sp[8];
    int b = blockIdx.x, tid = threadIdx.x;
    int lane = tid & 31, wid = tid >> 5;
    float p = 0.f;
    for (int u = tid; u < HID; u += 256) p += hbuf[(long long)b * HID + u] * out_W[u];
    #pragma unroll
    for (int off = 16; off > 0; off >>= 1) p += __shfl_down(p, off, 32);
    if (lane == 0) sp[wid] = p;
    __syncthreads();
    if (tid == 0) {
        float y = out_b[0];
        #pragma unroll
        for (int w = 0; w < 8; ++w) y += sp[w];
        pred[b] = 1.f / (1.f + __expf(-y));
        float lab = labels[b], it = is_train[b];
        float mv = fmaxf(-y, 0.f);
        float bce = y - y * lab + mv + __logf(__expf(-mv) + __expf(-y - mv));
        atomicAdd(&bce_acc[0], bce * it);
        atomicAdd(&bce_acc[1], it);
    }
}

__global__ void k_loss(const float* __restrict__ xl_num, const float* __restrict__ xl_den,
                       const float* __restrict__ bce_acc, float* __restrict__ loss_out) {
    if (threadIdx.x == 0 && blockIdx.x == 0) {
        float xl = 0.f;
        for (int t = 0; t < SEQ_LEN; ++t) xl += xl_num[t] / (xl_den[t] + 1e-5f);
        float yl = bce_acc[0] / (bce_acc[1] + 1e-5f);
        loss_out[0] = 0.3f * xl + 1.0f * yl;
    }
}

extern "C" void kernel_launch(void* const* d_in, const int* in_sizes, int n_in,
                              void* d_out, int out_size, void* d_ws, size_t ws_size,
                              hipStream_t stream) {
    const float* values   = (const float*)d_in[0];
    const float* masks    = (const float*)d_in[1];
    const float* deltas   = (const float*)d_in[2];
    const float* labels   = (const float*)d_in[3];
    const float* is_train = (const float*)d_in[4];
    const float* td_W     = (const float*)d_in[5];
    const float* td_b     = (const float*)d_in[6];
    const float* W_ih     = (const float*)d_in[7];
    const float* W_hh     = (const float*)d_in[8];
    const float* b_ih     = (const float*)d_in[9];
    const float* b_hh     = (const float*)d_in[10];
    const float* reg_W    = (const float*)d_in[11];
    const float* reg_b    = (const float*)d_in[12];
    const float* out_W    = (const float*)d_in[13];
    const float* out_b    = (const float*)d_in[14];

    char* ws = (char*)d_ws;
    size_t off = 0;
    auto alloc = [&](size_t bytes) -> void* {
        void* p = ws + off;
        off = (off + bytes + 255) & ~(size_t)255;
        return p;
    };
    __bf16* Wbuf = (__bf16*)alloc((size_t)GATES * KPAD * 2);  // ~9.2 MB (L2-resident)
    __bf16* Abuf = (__bf16*)alloc((size_t)BATCH * KPAD * 2);  // ~9.2 MB
    float*  hbuf  = (float*)alloc((size_t)BATCH * HID * 4);
    float*  cbuf  = (float*)alloc((size_t)BATCH * HID * 4);
    float*  bsum  = (float*)alloc((size_t)GATES * 4);
    float*  xl    = (float*)alloc(2 * SEQ_LEN * 4);           // num[48] | den[48]
    float*  bce   = (float*)alloc(2 * 4);

    float* out      = (float*)d_out;
    float* loss_out = out;
    float* pred     = out + 1;
    float* imput    = out + 1 + BATCH;

    long long nW = (long long)GATES * KPAD;
    long long nA = (long long)BATCH * KPAD;
    k_prep_w<<<(unsigned)((nW + 255) / 256), 256, 0, stream>>>(W_hh, W_ih, Wbuf);
    k_zero_state<<<(unsigned)((nA + 255) / 256), 256, 0, stream>>>(hbuf, cbuf, Abuf,
                                                                   b_ih, b_hh, bsum, xl, bce);

    dim3 ggrid(HID / TJ, BATCH / TM);   // (16, 32)
    for (int t = 0; t < SEQ_LEN; ++t) {
        k_decay_impute<<<BATCH, 256, 0, stream>>>(t, values, masks, deltas, td_W, td_b,
                                                  reg_W, reg_b, hbuf, Abuf, imput,
                                                  xl, xl + SEQ_LEN);
        k_gates_gemm_lstm<<<ggrid, 256, 0, stream>>>(Abuf, Wbuf, bsum, hbuf, cbuf);
    }
    k_out<<<BATCH, 256, 0, stream>>>(hbuf, out_W, out_b, labels, is_train, pred, bce);
    k_loss<<<1, 1, 0, stream>>>(xl, xl + SEQ_LEN, bce, loss_out);
}
